// AttStateEncoder_43327630082310
// MI455X (gfx1250) — matrix operand order repaired and checked by reading, not compiled, not run on hardware
//
#include <hip/hip_runtime.h>
#include <hip/hip_bf16.h>

typedef __attribute__((ext_vector_type(16))) __bf16 v16bf;
typedef __attribute__((ext_vector_type(8)))  float  v8f;

#define BH        (512*128)
#define DD        75
#define NTOK      32
#define SMDIM     32
#define HALFD     128
#define EMB       256
#define ROWS      128
#define THREADS   256
#define STG_STRIDE 107                    /* 75 state + 32 scores; 107%64=43, coprime -> no bank conflicts */
#define XA_STRIDE  264                    /* 256 + 8 pad bf16; row byte stride 528 (16B aligned) */
#define XA_BYTES   (ROWS*XA_STRIDE*2)
#define STG_BYTES  (ROWS*STG_STRIDE*4)
#define SMEM_BYTES (XA_BYTES + STG_BYTES)

union BFrag { v16bf v; uint4 q[2]; };

__device__ __forceinline__ unsigned short f2bf(float f) {
    unsigned int u = __float_as_uint(f);
    u += 0x7fffu + ((u >> 16) & 1u);      // round-to-nearest-even
    return (unsigned short)(u >> 16);
}

// ---------------- kernel 0: fuse_W f32 -> bf16 (row-major, 256x256) ----------------
__global__ void convW_kernel(const float* __restrict__ W, unsigned short* __restrict__ Wbf, int n) {
    int i = blockIdx.x * blockDim.x + threadIdx.x;
    if (i < n) Wbf[i] = f2bf(W[i]);
}

// ---------------- main fused encoder kernel ----------------
__global__ void __launch_bounds__(THREADS)
enc_kernel(const float* __restrict__ state,
           const float* __restrict__ token_W, const float* __restrict__ token_b,
           const float* __restrict__ query_W, const float* __restrict__ query_b,
           const float* __restrict__ p_rbs,  const float* __restrict__ p_vb,
           const float* __restrict__ p_ivb,
           const float* __restrict__ att_ln_w, const float* __restrict__ att_ln_b,
           const float* __restrict__ non_W,  const float* __restrict__ non_b,
           const float* __restrict__ spfb_W, const float* __restrict__ spfb_b,
           const float* __restrict__ p_gate,
           const float* __restrict__ spatial_W, const float* __restrict__ spatial_b,
           const unsigned short* __restrict__ Wbf,  /* bf16 fuse_W [256][256] */
           const float* __restrict__ fuse_b,
           const float* __restrict__ ln_w, const float* __restrict__ ln_b,
           float* __restrict__ out_x, float* __restrict__ out_tf) {
    extern __shared__ char smem[];
    unsigned short* xA = reinterpret_cast<unsigned short*>(smem);          // bf16 x [128][264]
    float* stage       = reinterpret_cast<float*>(smem + XA_BYTES);        // f32 [128][107]

    const int tid = threadIdx.x;

    // ---------- phase 0: coalesced stage of state rows into LDS ----------
    {
        const float* src = state + (size_t)blockIdx.x * ROWS * DD;
        for (int i = tid; i < ROWS * DD; i += THREADS) {
            int r = i / DD;
            int c = i - r * DD;
            stage[r * STG_STRIDE + c] = src[i];
        }
    }
    __syncthreads();

    // ---------- phase 1: per-row scalar pipeline (threads 0..127) ----------
    if (tid < ROWS) {
        const int t = tid;
        float* S = stage + t * STG_STRIDE;
        const int rowg = blockIdx.x * ROWS + t;

        float mn = 1e30f, sr = 0.f, vc = 0.f, dsum = 0.f, dmx = 0.f;
        for (int n = 0; n < NTOK; ++n) {
            float s = S[n], d = fabsf(S[32 + n]);
            mn = fminf(mn, s); sr += s;
            vc += (s < 0.95f) ? 1.f : 0.f;
            dsum += d; dmx = fmaxf(dmx, d);
        }
        float mean_r = sr * (1.f / 32.f);
        float vrat   = vc * (1.f / 32.f);
        float dmean  = dsum * (1.f / 32.f);
        float dyaw   = fminf(fabsf(S[72]), 1.f);
        float dmag   = fminf(dmean, 1.f);

        float na[17];
        #pragma unroll
        for (int k = 0; k < 11; ++k) na[k] = S[64 + k];
        na[11] = mn; na[12] = mean_r; na[13] = vrat; na[14] = dmean; na[15] = dmx; na[16] = dyaw;

        float q[SMDIM];
        #pragma unroll
        for (int s = 0; s < SMDIM; ++s) {
            float a = query_b[s];
            #pragma unroll
            for (int k = 0; k < 17; ++k) a += query_W[s * 17 + k] * na[k];
            q[s] = fmaxf(a, 0.f);
        }

        const float rbs = p_rbs[0], vb = p_vb[0], ivb = p_ivb[0];

        // token pass 1: scores (+ running max)
        float smax = -1e30f;
        for (int n = 0; n < NTOK; ++n) {
            float sec = S[n], dif = S[32 + n];
            float val = (sec < 0.95f) ? 1.f : 0.f;
            float sc = 0.f;
            #pragma unroll
            for (int s = 0; s < SMDIM; ++s) {
                float e = fmaxf(token_W[s * 3 + 0] * sec + token_W[s * 3 + 1] * dif +
                                token_W[s * 3 + 2] * val + token_b[s], 0.f);
                sc += e * q[s];
            }
            sc = sc * 0.17677669529663687f + rbs * (1.f - sec) + vb * val + ivb * (1.f - val);
            S[75 + n] = sc;
            smax = fmaxf(smax, sc);
        }

        // token pass 2: softmax-weighted context
        float ctx[SMDIM];
        #pragma unroll
        for (int s = 0; s < SMDIM; ++s) ctx[s] = 0.f;
        float wsum = 0.f;
        for (int n = 0; n < NTOK; ++n) {
            float sec = S[n], dif = S[32 + n];
            float val = (sec < 0.95f) ? 1.f : 0.f;
            float w = __expf(S[75 + n] - smax);
            wsum += w;
            #pragma unroll
            for (int s = 0; s < SMDIM; ++s) {
                float e = fmaxf(token_W[s * 3 + 0] * sec + token_W[s * 3 + 1] * dif +
                                token_W[s * 3 + 2] * val + token_b[s], 0.f);
                ctx[s] += w * e;
            }
        }
        float inv = 1.f / wsum;
        float m = 0.f;
        #pragma unroll
        for (int s = 0; s < SMDIM; ++s) { ctx[s] *= inv; m += ctx[s]; }
        m *= (1.f / 32.f);
        float va = 0.f;
        #pragma unroll
        for (int s = 0; s < SMDIM; ++s) { float d = ctx[s] - m; va += d * d; }
        va *= (1.f / 32.f);
        float rstd = rsqrtf(va + 1e-5f);
        float cn[SMDIM];
        #pragma unroll
        for (int s = 0; s < SMDIM; ++s) cn[s] = (ctx[s] - m) * rstd * att_ln_w[s] + att_ln_b[s];

        float alpha = 1.f / (1.f + __expf(-p_gate[0]));

        unsigned short* xr = xA + t * XA_STRIDE;
        for (int c = 0; c < HALFD; ++c) {
            float ne = non_b[c];
            #pragma unroll
            for (int k = 0; k < 17; ++k) ne += non_W[c * 17 + k] * na[k];
            ne = fmaxf(ne, 0.f);
            float sa = spatial_b[c];
            #pragma unroll
            for (int s = 0; s < SMDIM; ++s) sa += spatial_W[c * 32 + s] * cn[s];
            sa = fmaxf(sa, 0.f);
            float sf = fmaxf(spfb_b[c] + spfb_W[c * 3 + 0] * mn + spfb_W[c * 3 + 1] * mean_r +
                             spfb_W[c * 3 + 2] * vrat, 0.f);
            float sp = alpha * sa + (1.f - alpha) * sf;
            xr[c]         = f2bf(ne);
            xr[HALFD + c] = f2bf(sp);
        }

        out_tf[(size_t)rowg * 2 + 0] = dmag;
        out_tf[(size_t)rowg * 2 + 1] = dyaw;
    }
    __syncthreads();

    // ---------- phase 2: fuse GEMM via bf16 WMMA; each wave = one 16-row M-tile ----------
    const int lane = tid & 31;
    const int mt   = tid >> 5;     // wave id: 0..7
    const int ln16 = lane & 15;
    const int hi   = lane >> 4;    // 0 or 1

    v8f acc[16];
    #pragma unroll
    for (int nt = 0; nt < 16; ++nt)
        #pragma unroll
        for (int j = 0; j < 8; ++j) acc[nt][j] = 0.f;

    const unsigned short* xrow = xA + (mt * 16 + ln16) * XA_STRIDE;  // A: M = lane%16
    #pragma unroll
    for (int nt = 0; nt < 16; ++nt) {
        const unsigned short* wrow = Wbf + (size_t)(nt * 16 + ln16) * EMB + hi * 16; // B: N = lane%16
        #pragma unroll
        for (int kt = 0; kt < 8; ++kt) {
            BFrag a, b;
            // A frag: 16 consecutive K at kt*32 + 16*hi (matches B order)
            a.q[0] = *reinterpret_cast<const uint4*>(xrow + kt * 32 + hi * 16);
            a.q[1] = *reinterpret_cast<const uint4*>(xrow + kt * 32 + hi * 16 + 8);
            // B frag: 16 consecutive K at kt*32 + 16*hi for this lane's column of fuse_W
            const uint4* pb = reinterpret_cast<const uint4*>(wrow + kt * 32);
            b.q[0] = pb[0];
            b.q[1] = pb[1];
            acc[nt] = __builtin_amdgcn_wmma_f32_16x16x32_bf16(
                false, a.v, false, b.v, (short)0, acc[nt], false, false);
        }
    }

    // ---------- epilogue: bias + relu + LayerNorm fused in registers ----------
    float fb[16], lw[16], lb[16];
    #pragma unroll
    for (int nt = 0; nt < 16; ++nt) {
        fb[nt] = fuse_b[nt * 16 + ln16];
        lw[nt] = ln_w[nt * 16 + ln16];
        lb[nt] = ln_b[nt * 16 + ln16];
    }
    #pragma unroll
    for (int j = 0; j < 8; ++j) {
        float s1 = 0.f, s2 = 0.f;
        #pragma unroll
        for (int nt = 0; nt < 16; ++nt) {
            float v = fmaxf(acc[nt][j] + fb[nt], 0.f);
            s1 += v; s2 += v * v;
        }
        // reduce across the 16-lane group holding this row (C/D: N = lane%16)
        s1 += __shfl_xor(s1, 1); s1 += __shfl_xor(s1, 2);
        s1 += __shfl_xor(s1, 4); s1 += __shfl_xor(s1, 8);
        s2 += __shfl_xor(s2, 1); s2 += __shfl_xor(s2, 2);
        s2 += __shfl_xor(s2, 4); s2 += __shfl_xor(s2, 8);
        float mean = s1 * (1.f / 256.f);
        float var  = s2 * (1.f / 256.f) - mean * mean;
        float rs   = rsqrtf(var + 1e-5f);
        int rowg = blockIdx.x * ROWS + mt * 16 + hi * 8 + j;   // C/D: M = j + 8*(lane>=16)
        float* orow = out_x + (size_t)rowg * EMB + ln16;
        #pragma unroll
        for (int nt = 0; nt < 16; ++nt) {
            float v = fmaxf(acc[nt][j] + fb[nt], 0.f);
            orow[nt * 16] = (v - mean) * rs * lw[nt] + lb[nt];
        }
    }
}

extern "C" void kernel_launch(void* const* d_in, const int* in_sizes, int n_in,
                              void* d_out, int out_size, void* d_ws, size_t ws_size,
                              hipStream_t stream) {
    (void)in_sizes; (void)n_in; (void)out_size; (void)ws_size;
    const float* state     = (const float*)d_in[0];
    const float* token_W   = (const float*)d_in[1];
    const float* token_b   = (const float*)d_in[2];
    const float* query_W   = (const float*)d_in[3];
    const float* query_b   = (const float*)d_in[4];
    const float* rbs       = (const float*)d_in[5];
    const float* vb        = (const float*)d_in[6];
    const float* ivb       = (const float*)d_in[7];
    const float* att_ln_w  = (const float*)d_in[8];
    const float* att_ln_b  = (const float*)d_in[9];
    const float* non_W     = (const float*)d_in[10];
    const float* non_b     = (const float*)d_in[11];
    const float* spfb_W    = (const float*)d_in[12];
    const float* spfb_b    = (const float*)d_in[13];
    const float* gate      = (const float*)d_in[14];
    const float* spatial_W = (const float*)d_in[15];
    const float* spatial_b = (const float*)d_in[16];
    const float* fuse_W    = (const float*)d_in[17];
    const float* fuse_b    = (const float*)d_in[18];
    const float* ln_w      = (const float*)d_in[19];
    const float* ln_b      = (const float*)d_in[20];

    unsigned short* Wbf = (unsigned short*)d_ws;           // 256*256 bf16 = 128 KB
    float* out_x  = (float*)d_out;
    float* out_tf = out_x + (size_t)BH * EMB;

    convW_kernel<<<dim3((EMB * EMB) / 256), dim3(256), 0, stream>>>(fuse_W, Wbf, EMB * EMB);

    enc_kernel<<<dim3(BH / ROWS), dim3(THREADS), SMEM_BYTES, stream>>>(
        state, token_W, token_b, query_W, query_b, rbs, vb, ivb,
        att_ln_w, att_ln_b, non_W, non_b, spfb_W, spfb_b, gate,
        spatial_W, spatial_b, Wbf, fuse_b, ln_w, ln_b, out_x, out_tf);
}
